// RawCrossAttention_47802986004881
// MI455X (gfx1250) — matrix operand order repaired and checked by reading, not compiled, not run on hardware
//
#include <hip/hip_runtime.h>
#include <hip/hip_bf16.h>
#include <math.h>

// Problem dims (fixed by the reference)
#define B_      2
#define CH      1024
#define TT      2048
#define SS      2048
#define NHEADS  16
#define HD      64
#define NEGV    1000000.0f

typedef __bf16 bf16_t;
typedef __attribute__((ext_vector_type(16))) __bf16 v16bf;
typedef __attribute__((ext_vector_type(8)))  float  v8f;
typedef int v4i_ __attribute__((vector_size(16)));

struct alignas(16) bf16x8 { bf16_t v[8]; };

// Workspace: six bf16 planes of B*H*2048*64 halves each (total 50.3 MB):
//   0: Q_hi [b][h][t][d]   1: Q_lo    (A-fragment layout rows)
//   2: K_hi [b][h][s][d]   3: K_lo    (B^T-fragment layout rows)
//   4: V_hi [b][h][d][s]   5: V_lo    (B^T layout for P·V)
#define PLANE ((size_t)B_ * NHEADS * TT * HD)

// ---- CDNA5 async LDS copy path (ASYNCcnt) with portable fallback ----
#if defined(__AMDGCN__) && __has_builtin(__builtin_amdgcn_global_load_async_to_lds_b128)
#define ASYNC_LDS 1
#else
#define ASYNC_LDS 0
#endif

__device__ __forceinline__ void cp16_g2l(bf16_t* lds, const bf16_t* g) {
#if ASYNC_LDS
  bf16_t* gg = const_cast<bf16_t*>(g);
  __builtin_amdgcn_global_load_async_to_lds_b128(
      (__attribute__((address_space(1))) v4i_*)gg,
      (__attribute__((address_space(3))) v4i_*)lds, 0, 0);
#else
  *(bf16x8*)lds = *(const bf16x8*)g;
#endif
}

__device__ __forceinline__ void wait_async() {
#if ASYNC_LDS
#if __has_builtin(__builtin_amdgcn_s_wait_asynccnt)
  __builtin_amdgcn_s_wait_asynccnt(0);
#else
  asm volatile("s_wait_asynccnt 0" ::: "memory");
#endif
#endif
}

// ---- CDNA5 WMMA fragment loaders (ISA 7.12.2 layouts, wave32) ----
// A (16xK bf16, M rows): lanes 0-15 row m, halves K=[k0+0..7] then [k0+16..23];
// lanes 16-31 row m, halves K=[k0+8..15] then [k0+24..31].
__device__ __forceinline__ v16bf frag_a(const bf16_t* base, int m0, int stride,
                                        int k0, int lane) {
  const bf16_t* p = base + (size_t)(m0 + (lane & 15)) * stride + k0 + ((lane >> 4) << 3);
  v16bf a;
  ((bf16x8*)&a)[0] = *(const bf16x8*)(p);
  ((bf16x8*)&a)[1] = *(const bf16x8*)(p + 16);
  return a;
}
// B (Kx16 bf16) stored transposed in LDS as [n][k]: lane n holds column n,
// lanes 0-15 K=[k0+0..7] then [k0+16..23]; lanes 16-31 K=[k0+8..15] then [k0+24..31].
__device__ __forceinline__ v16bf frag_b(const bf16_t* base, int n0, int stride,
                                        int k0, int lane) {
  const bf16_t* p = base + (size_t)(n0 + (lane & 15)) * stride + k0 + ((lane >> 4) << 3);
  v16bf b;
  ((bf16x8*)&b)[0] = *(const bf16x8*)(p);
  ((bf16x8*)&b)[1] = *(const bf16x8*)(p + 16);
  return b;
}

#define WMMA(c, a, bb)                                                         \
  (c) = __builtin_amdgcn_wmma_f32_16x16x32_bf16(false, (a), false, (bb),       \
                                                (short)0, (c), false, false)

__device__ __forceinline__ void split_bf16(float x, bf16_t& hi, bf16_t& lo) {
  hi = (bf16_t)x;
  lo = (bf16_t)(x - (float)hi);
}

// =====================================================================
// Kernel 1: fused QKV projection, bf16x3 split GEMM (128x128 tile, K-step 32).
// Epilogue writes bf16 hi/lo planes pre-transposed for kernel 2's LDS layouts.
// =====================================================================
__global__ __launch_bounds__(256) void qkv_proj_kernel(
    const float* __restrict__ target, const float* __restrict__ source,
    const float* __restrict__ Wq, const float* __restrict__ bq,
    const float* __restrict__ Wk, const float* __restrict__ bk,
    const float* __restrict__ Wv, const float* __restrict__ bv,
    bf16_t* __restrict__ wsb) {
  const int z = blockIdx.z, proj = z >> 1, b = z & 1;
  const float* W; const float* bias; const float* X;
  if (proj == 0)      { W = Wq; bias = bq; X = target; }
  else if (proj == 1) { W = Wk; bias = bk; X = source; }
  else                { W = Wv; bias = bv; X = source; }
  X += (size_t)b * CH * TT;

  const int o0 = blockIdx.y * 128;
  const int l0 = blockIdx.x * 128;
  const int tid = threadIdx.x, lane = tid & 31, wid = tid >> 5;
  const int wm = wid & 1;   // 2 wave rows  (64 M each == one head)
  const int wn = wid >> 1;  // 4 wave cols  (32 N each)

  __shared__ bf16_t Ahi[128][32], Alo[128][32];  // W tile, [m][k]
  __shared__ bf16_t Bhi[128][32], Blo[128][32];  // X tile transposed, [n=l][k]

  v8f acc[4][2];
  v8f zv = {};
#pragma unroll
  for (int mt = 0; mt < 4; ++mt)
#pragma unroll
    for (int nt = 0; nt < 2; ++nt) acc[mt][nt] = zv;

  for (int k0 = 0; k0 < CH; k0 += 32) {
    __syncthreads();
    {  // stage W tile: 128 rows x 32 k, row-major
      int r = tid >> 1, cg = (tid & 1) * 16;
      const float4* src = (const float4*)(W + (size_t)(o0 + r) * CH + k0 + cg);
#pragma unroll
      for (int j = 0; j < 4; ++j) {
        float4 f = src[j];
        int c = cg + j * 4;
        split_bf16(f.x, Ahi[r][c + 0], Alo[r][c + 0]);
        split_bf16(f.y, Ahi[r][c + 1], Alo[r][c + 1]);
        split_bf16(f.z, Ahi[r][c + 2], Alo[r][c + 2]);
        split_bf16(f.w, Ahi[r][c + 3], Alo[r][c + 3]);
      }
    }
    {  // stage X tile transposed: [l][k]
      int k = tid >> 3, lg = (tid & 7) * 16;
      const float4* src = (const float4*)(X + (size_t)(k0 + k) * TT + l0 + lg);
#pragma unroll
      for (int j = 0; j < 4; ++j) {
        float4 f = src[j];
        int l = lg + j * 4;
        split_bf16(f.x, Bhi[l + 0][k], Blo[l + 0][k]);
        split_bf16(f.y, Bhi[l + 1][k], Blo[l + 1][k]);
        split_bf16(f.z, Bhi[l + 2][k], Blo[l + 2][k]);
        split_bf16(f.w, Bhi[l + 3][k], Blo[l + 3][k]);
      }
    }
    __syncthreads();

    v16bf bh[2], bl[2];
#pragma unroll
    for (int nt = 0; nt < 2; ++nt) {
      bh[nt] = frag_b(&Bhi[0][0], wn * 32 + nt * 16, 32, 0, lane);
      bl[nt] = frag_b(&Blo[0][0], wn * 32 + nt * 16, 32, 0, lane);
    }
#pragma unroll
    for (int mt = 0; mt < 4; ++mt) {
      v16bf ah = frag_a(&Ahi[0][0], wm * 64 + mt * 16, 32, 0, lane);
      v16bf al = frag_a(&Alo[0][0], wm * 64 + mt * 16, 32, 0, lane);
#pragma unroll
      for (int nt = 0; nt < 2; ++nt) {
        WMMA(acc[mt][nt], ah, bh[nt]);  // hi*hi
        WMMA(acc[mt][nt], ah, bl[nt]);  // hi*lo
        WMMA(acc[mt][nt], al, bh[nt]);  // lo*hi
      }
    }
  }

  // ---- epilogue: +bias, split to bf16 hi/lo planes ----
  // wave's o-range o0+wm*64..+63 is exactly one head hw, d = local o.
  const int hw = (o0 + wm * 64) >> 6;
  const size_t HO = ((size_t)b * NHEADS + hw) * ((size_t)TT * HD);

  if (proj <= 1) {
    // Q/K planes: [t|s][d] rows -> pack 8 consecutive d into one b128 store.
    bf16_t* PH = wsb + (size_t)(proj * 2) * PLANE + HO;
    bf16_t* PL = PH + PLANE;
#pragma unroll
    for (int mt = 0; mt < 4; ++mt)
#pragma unroll
      for (int nt = 0; nt < 2; ++nt) {
        int t = l0 + wn * 32 + nt * 16 + (lane & 15);
        int dbase = mt * 16 + ((lane >> 4) << 3);
        bf16x8 hi8, lo8;
#pragma unroll
        for (int i = 0; i < 8; ++i) {
          float v = acc[mt][nt][i] + bias[hw * 64 + dbase + i];
          hi8.v[i] = (bf16_t)v;
          lo8.v[i] = (bf16_t)(v - (float)hi8.v[i]);
        }
        *(bf16x8*)(PH + (size_t)t * HD + dbase) = hi8;
        *(bf16x8*)(PL + (size_t)t * HD + dbase) = lo8;
      }
  } else {
    // V planes: [d][s], s across lanes (adjacent-lane coalescing).
    bf16_t* PH = wsb + 4 * PLANE + HO;
    bf16_t* PL = PH + PLANE;
#pragma unroll
    for (int mt = 0; mt < 4; ++mt)
#pragma unroll
      for (int nt = 0; nt < 2; ++nt) {
        int s = l0 + wn * 32 + nt * 16 + (lane & 15);
        int dbase = mt * 16 + ((lane >> 4) << 3);
#pragma unroll
        for (int i = 0; i < 8; ++i) {
          int d = dbase + i;
          float v = acc[mt][nt][i] + bias[hw * 64 + d];
          bf16_t hi, lo;
          split_bf16(v, hi, lo);
          PH[(size_t)d * SS + s] = hi;
          PL[(size_t)d * SS + s] = lo;
        }
      }
  }
}

// =====================================================================
// Kernel 2: flash cross-attention.  Block = (b, h, 64 t-rows); 4 waves x 16 rows.
// K/V tiles double-buffered via GLOBAL_LOAD_ASYNC_TO_LDS (ASYNCcnt) so copies
// overlap the WMMA stream; online softmax in fp32 VALU co-executes with WMMA.
// =====================================================================
__global__ __launch_bounds__(128) void attn_kernel(
    const bf16_t* __restrict__ wsb,
    const float* __restrict__ target_mask, const float* __restrict__ source_mask,
    float* __restrict__ out) {
  const int t0 = blockIdx.x * 64;
  const int h  = blockIdx.y;
  const int b  = blockIdx.z;
  const int tid = threadIdx.x, lane = tid & 31, wid = tid >> 5;
  const size_t HO = ((size_t)b * NHEADS + h) * ((size_t)TT * HD);
  const bf16_t* QH = wsb + 0 * PLANE + HO;
  const bf16_t* QL = wsb + 1 * PLANE + HO;
  const bf16_t* KH = wsb + 2 * PLANE + HO;
  const bf16_t* KL = wsb + 3 * PLANE + HO;
  const bf16_t* VH = wsb + 4 * PLANE + HO;
  const bf16_t* VL = wsb + 5 * PLANE + HO;

  __shared__ bf16_t Qh[64][64], Ql[64][64];        // A layout [t_local][d]
  __shared__ bf16_t Kh[2][64][64], Kl[2][64][64];  // B^T layout [s_local][d], 2 bufs
  __shared__ bf16_t Vh[2][64][64], Vl[2][64][64];  // B^T layout [d][s_local], 2 bufs
  __shared__ bf16_t Ph[4][16][64], Pl[4][16][64];  // per-wave P, A layout

  // issue one K/V tile (hi+lo): 16B chunks, tiles are contiguous (K) or
  // 128B-row-strided (V) in the ws planes.
  auto stage_kv = [&](int s0, int buf) {
#pragma unroll
    for (int p = 0; p < 4; ++p) {
      int c = p * 128 + tid;  // chunk id, 512 x 16B per 8KB tile
      cp16_g2l(&Kh[buf][0][0] + c * 8, KH + (size_t)s0 * HD + c * 8);
      cp16_g2l(&Kl[buf][0][0] + c * 8, KL + (size_t)s0 * HD + c * 8);
      int d = c >> 3, j = c & 7;
      cp16_g2l(&Vh[buf][0][0] + c * 8, VH + (size_t)d * SS + s0 + j * 8);
      cp16_g2l(&Vl[buf][0][0] + c * 8, VL + (size_t)d * SS + s0 + j * 8);
    }
  };

  // prologue: Q tile (contiguous 8KB hi + 8KB lo) + first K/V tile
#pragma unroll
  for (int p = 0; p < 4; ++p) {
    int c = p * 128 + tid;
    cp16_g2l(&Qh[0][0] + c * 8, QH + (size_t)t0 * HD + c * 8);
    cp16_g2l(&Ql[0][0] + c * 8, QL + (size_t)t0 * HD + c * 8);
  }
  stage_kv(0, 0);

  v8f oacc[4];
  v8f zv = {};
#pragma unroll
  for (int dn = 0; dn < 4; ++dn) oacc[dn] = zv;
  float mst[8], lst[8];
#pragma unroll
  for (int i = 0; i < 8; ++i) { mst[i] = -3.0e38f; lst[i] = 0.0f; }

  for (int s0 = 0; s0 < SS; s0 += 64) {
    const int buf = (s0 >> 6) & 1;
    wait_async();     // my async copies for `buf` (and Q) done
    __syncthreads();  // everyone's copies done; prev reads of buf^1 done
    if (s0 + 64 < SS) stage_kv(s0 + 64, buf ^ 1);  // prefetch behind compute

    // ---- scores: 16(t) x 64(s) per wave, K=d=64 in two 32-steps ----
    v8f sc[4];
#pragma unroll
    for (int nt = 0; nt < 4; ++nt) sc[nt] = zv;
#pragma unroll
    for (int k0 = 0; k0 < HD; k0 += 32) {
      v16bf qh = frag_a(&Qh[0][0], wid * 16, 64, k0, lane);
      v16bf ql = frag_a(&Ql[0][0], wid * 16, 64, k0, lane);
#pragma unroll
      for (int nt = 0; nt < 4; ++nt) {
        v16bf kh = frag_b(&Kh[buf][0][0], nt * 16, 64, k0, lane);
        v16bf kl = frag_b(&Kl[buf][0][0], nt * 16, 64, k0, lane);
        WMMA(sc[nt], qh, kh);
        WMMA(sc[nt], qh, kl);
        WMMA(sc[nt], ql, kh);
      }
    }

    // ---- mask, scale, online softmax (row = (i, half); cols across 16 lanes) ----
    float msk[4];
#pragma unroll
    for (int nt = 0; nt < 4; ++nt)
      msk[nt] = source_mask[(size_t)b * SS + s0 + nt * 16 + (lane & 15)];

#pragma unroll
    for (int i = 0; i < 8; ++i) {
      float rmax = -NEGV;
#pragma unroll
      for (int nt = 0; nt < 4; ++nt) {
        float x = (msk[nt] != 0.0f) ? sc[nt][i] * 0.125f : -NEGV;
        sc[nt][i] = x;
        rmax = fmaxf(rmax, x);
      }
      rmax = fmaxf(rmax, __shfl_xor(rmax, 1, 32));
      rmax = fmaxf(rmax, __shfl_xor(rmax, 2, 32));
      rmax = fmaxf(rmax, __shfl_xor(rmax, 4, 32));
      rmax = fmaxf(rmax, __shfl_xor(rmax, 8, 32));
      float mnew  = fmaxf(mst[i], rmax);
      float alpha = expf(mst[i] - mnew);
      float psum  = 0.0f;
#pragma unroll
      for (int nt = 0; nt < 4; ++nt) {
        float p = (msk[nt] != 0.0f) ? expf(sc[nt][i] - mnew) : 0.0f;  // amap*mask
        sc[nt][i] = p;
        psum += p;
      }
      psum += __shfl_xor(psum, 1, 32);
      psum += __shfl_xor(psum, 2, 32);
      psum += __shfl_xor(psum, 4, 32);
      psum += __shfl_xor(psum, 8, 32);
      lst[i] = lst[i] * alpha + psum;
      mst[i] = mnew;
#pragma unroll
      for (int dn = 0; dn < 4; ++dn) oacc[dn][i] *= alpha;
    }

    // ---- C-layout P -> A-layout via per-wave LDS (hi/lo split) ----
#pragma unroll
    for (int nt = 0; nt < 4; ++nt)
#pragma unroll
      for (int i = 0; i < 8; ++i) {
        int r = i + ((lane >> 4) << 3);
        int c = nt * 16 + (lane & 15);
        bf16_t hi, lo;
        split_bf16(sc[nt][i], hi, lo);
        Ph[wid][r][c] = hi;
        Pl[wid][r][c] = lo;
      }

    // ---- O += P · V (K=s=64 in two 32-steps) ----
#pragma unroll
    for (int k0 = 0; k0 < 64; k0 += 32) {
      v16bf ph = frag_a(&Ph[wid][0][0], 0, 64, k0, lane);
      v16bf pl = frag_a(&Pl[wid][0][0], 0, 64, k0, lane);
#pragma unroll
      for (int dn = 0; dn < 4; ++dn) {
        v16bf vh = frag_b(&Vh[buf][0][0], dn * 16, 64, k0, lane);
        v16bf vl = frag_b(&Vl[buf][0][0], dn * 16, 64, k0, lane);
        WMMA(oacc[dn], ph, vh);
        WMMA(oacc[dn], ph, vl);
        WMMA(oacc[dn], pl, vh);
      }
    }
  }

  // ---- normalize, apply target mask, store [B, H*D, T] ----
#pragma unroll
  for (int i = 0; i < 8; ++i) {
    int t = t0 + wid * 16 + i + ((lane >> 4) << 3);
    float tm = target_mask[(size_t)b * TT + t];
    float l  = lst[i];
    float inv = (tm != 0.0f && l > 0.0f) ? 1.0f / l : 0.0f;
#pragma unroll
    for (int dn = 0; dn < 4; ++dn) {
      int d = dn * 16 + (lane & 15);
      out[(size_t)b * CH * TT + (size_t)(h * HD + d) * TT + t] = oacc[dn][i] * inv;
    }
  }
}

// =====================================================================
extern "C" void kernel_launch(void* const* d_in, const int* in_sizes, int n_in,
                              void* d_out, int out_size, void* d_ws, size_t ws_size,
                              hipStream_t stream) {
  (void)in_sizes; (void)n_in; (void)out_size; (void)ws_size;
  const float* target = (const float*)d_in[0];
  const float* source = (const float*)d_in[1];
  const float* tmask  = (const float*)d_in[2];
  const float* smask  = (const float*)d_in[3];
  const float* Wq = (const float*)d_in[4];
  const float* bq = (const float*)d_in[5];
  const float* Wk = (const float*)d_in[6];
  const float* bk = (const float*)d_in[7];
  const float* Wv = (const float*)d_in[8];
  const float* bv = (const float*)d_in[9];
  float* out  = (float*)d_out;
  bf16_t* wsb = (bf16_t*)d_ws;  // needs 6 planes * 8.4 MB = 50.3 MB

  dim3 g1(TT / 128, CH / 128, 3 * B_);  // (16, 8, 6)
  qkv_proj_kernel<<<g1, 256, 0, stream>>>(target, source, Wq, bq, Wk, bk, Wv, bv, wsb);

  dim3 g2(TT / 64, NHEADS, B_);         // (32, 16, 2)
  attn_kernel<<<g2, 128, 0, stream>>>(wsb, tmask, smask, out);
}
